// gaussian_nonlocal_50886772523206
// MI455X (gfx1250) — matrix operand order repaired and checked by reading, not compiled, not run on hardware
//
#include <hip/hip_runtime.h>
#include <hip/hip_bf16.h>

#define NB 4
#define C_ 256
#define HW 4096
#define CB 128
#define PLD 40    // padded LDS row (ushorts) for P' tile; 80B, 16B-aligned
#define XLD 264   // padded LDS row (ushorts) for Xq tile; 528B, 16B-aligned

typedef unsigned short ushortT;
typedef __attribute__((ext_vector_type(16))) __bf16 v16bf;
typedef __attribute__((ext_vector_type(8)))  __bf16 v8bf;
typedef __attribute__((ext_vector_type(8)))  float  v8f;
typedef __attribute__((ext_vector_type(4)))  int    v4i;

#define GLOBAL_AS __attribute__((address_space(1)))
#define LDS_AS    __attribute__((address_space(3)))

#if defined(__AMDGCN__) && __has_builtin(__builtin_amdgcn_global_load_async_to_lds_b128)
#define HAVE_ASYNC 1
#else
#define HAVE_ASYNC 0
#endif

__device__ __forceinline__ void wait_async0() {
#if HAVE_ASYNC
#if __has_builtin(__builtin_amdgcn_s_wait_asynccnt)
    __builtin_amdgcn_s_wait_asynccnt(0);
#else
    asm volatile("s_wait_asynccnt 0x0" ::: "memory");
#endif
#endif
}

__device__ __forceinline__ ushortT f2bf(float f) {
    unsigned int u = __float_as_uint(f);
    unsigned int r = (u + 0x7FFFu + ((u >> 16) & 1u)) >> 16;   // round-to-nearest-even
    return (ushortT)r;
}

__device__ __forceinline__ v16bf make16(v8bf a, v8bf b) {
    return __builtin_shufflevector(a, b, 0,1,2,3,4,5,6,7,8,9,10,11,12,13,14,15);
}

// A-matrix fragment (16x32 bf16, M x K), source row-major [row][k], ld elems/row.
// Lane<16: row=row0+lane, K k0+{0..7},{16..23}; lane>=16: K k0+{8..15},{24..31}.
__device__ __forceinline__ v16bf load_a_frag(const ushortT* base, int ld, int row0, int k0, int lane) {
    int half = lane >> 4;
    const ushortT* p = base + (size_t)(row0 + (lane & 15)) * ld + k0 + half * 8;
    v8bf lo = *reinterpret_cast<const v8bf*>(p);
    v8bf hi = *reinterpret_cast<const v8bf*>(p + 16);
    return make16(lo, hi);
}

// B-matrix fragment (32x16 bf16, K x N) read from *transposed* storage base[n][k].
// Lane<16: n=n0+lane, K k0+{0..7},{16..23}; lane>=16: K k0+{8..15},{24..31}.
__device__ __forceinline__ v16bf load_bT_frag(const ushortT* base, int ld, int k0, int n0, int lane) {
    const ushortT* p = base + (size_t)(n0 + (lane & 15)) * ld + k0 + ((lane >> 4) << 3);
    v8bf lo = *reinterpret_cast<const v8bf*>(p);
    v8bf hi = *reinterpret_cast<const v8bf*>(p + 16);
    return make16(lo, hi);
}

__device__ __forceinline__ v8f wmma_bf16(v16bf a, v16bf b, v8f c) {
    return __builtin_amdgcn_wmma_f32_16x16x32_bf16(false, a, false, b, (short)0, c, false, false);
}

// Stage rows [q0, q0+32) of xt (256 bf16 cols each) into LDS (padded ld XLD).
// 256 threads x 4 x 16B = 16KB; async (ASYNCcnt) when available.
__device__ __forceinline__ void fill_xq(const ushortT* __restrict__ xtb, int q0,
                                        ushortT* lds, int tid) {
#pragma unroll
    for (int j = 0; j < 4; ++j) {
        int seg = tid + j * 256;                 // 16B segment index, lane-consecutive
        int r = seg >> 5, c8 = (seg & 31) * 8;
        const ushortT* gsrc = xtb + (size_t)(q0 + r) * C_ + c8;
        ushortT* ldst = lds + r * XLD + c8;
#if HAVE_ASYNC
        __builtin_amdgcn_global_load_async_to_lds_b128(
            (GLOBAL_AS v4i*)const_cast<ushortT*>(gsrc),
            (LDS_AS v4i*)ldst, 0, 0);
#else
        *reinterpret_cast<v8bf*>(ldst) = *reinterpret_cast<const v8bf*>(gsrc);
#endif
    }
}

// ---------------- prep: x [n][C][HW] f32 -> xt [n][HW][C] bf16 ----------------------
__global__ void k_xt(const float* __restrict__ x, ushortT* __restrict__ xt) {
    __shared__ float tile[32][33];
    int b = blockIdx.z;
    int p0 = blockIdx.x * 32;
    int c0 = blockIdx.y * 32;
    const float* xb = x + (size_t)b * C_ * HW;
    ushortT* xtb = xt + (size_t)b * HW * C_;
    int tx = threadIdx.x, ty = threadIdx.y;            // 32 x 8
#pragma unroll
    for (int j = 0; j < 32; j += 8)
        tile[ty + j][tx] = xb[(size_t)(c0 + ty + j) * HW + p0 + tx];
    __syncthreads();
#pragma unroll
    for (int j = 0; j < 32; j += 8)
        xtb[(size_t)(p0 + ty + j) * C_ + c0 + tx] = f2bf(tile[tx][ty + j]);
}

__global__ void k_cvt(const float* __restrict__ in, ushortT* __restrict__ out, int n) {
    int i = blockIdx.x * blockDim.x + threadIdx.x;
    if (i < n) out[i] = f2bf(in[i]);
}

// ---------------- g[cb][q] = Wg x + bg  (M=Cb, N=HW, K=C); 2 q-tiles/wave ------------
__global__ void k_g(const ushortT* __restrict__ xt, const ushortT* __restrict__ wgb,
                    const float* __restrict__ bg, ushortT* __restrict__ g) {
    int b = blockIdx.y;
    int q0 = blockIdx.x * 32;
    int wave = threadIdx.x >> 5, lane = threadIdx.x & 31;
    int cb0 = wave * 16;
    const ushortT* xtb = xt + (size_t)b * HW * C_;
    ushortT* gb = g + (size_t)b * CB * HW;
    v8f acc0 = {}, acc1 = {};
#pragma unroll
    for (int kc = 0; kc < C_; kc += 32) {
        v16bf a  = load_a_frag(wgb, C_, cb0, kc, lane);
        v16bf b0 = load_bT_frag(xtb, C_, kc, q0, lane);
        v16bf b1 = load_bT_frag(xtb, C_, kc, q0 + 16, lane);
        acc0 = wmma_bf16(a, b0, acc0);                 // two independent chains:
        acc1 = wmma_bf16(a, b1, acc1);                 // hides WMMA RAW-hazard slack
    }
    int col = lane & 15, half = lane >> 4;
#pragma unroll
    for (int v = 0; v < 8; ++v) {
        int cb = cb0 + v + 8 * half;
        float bias = bg[cb];
        gb[(size_t)cb * HW + q0 + col]      = f2bf(acc0[v] + bias);
        gb[(size_t)cb * HW + q0 + 16 + col] = f2bf(acc1[v] + bias);
    }
}

// ---------------- stats: c[q] = max_p f[q,p] + log(sum_p exp(f[q,p]-max)) ------------
__global__ void k_stats(const ushortT* __restrict__ xt, float* __restrict__ cst) {
    int b = blockIdx.y;
    int q0 = blockIdx.x * 16;
    int wave = threadIdx.x >> 5, lane = threadIdx.x & 31;
    const ushortT* xtb = xt + (size_t)b * HW * C_;
    float m[8], s[8];
#pragma unroll
    for (int v = 0; v < 8; ++v) { m[v] = -3.0e38f; s[v] = 0.f; }

    for (int pblk = 0; pblk < HW; pblk += 256) {       // two column tiles per iter
        int n0 = pblk + wave * 16;
        int n1 = n0 + 128;
        __builtin_prefetch(xtb + (size_t)(pblk + 256 < HW ? n0 + 256 : n0) * C_, 0, 1);
        v8f acc0 = {}, acc1 = {};
#pragma unroll
        for (int kc = 0; kc < C_; kc += 32) {
            v16bf a  = load_a_frag(xtb, C_, q0, kc, lane);
            v16bf b0 = load_bT_frag(xtb, C_, kc, n0, lane);
            v16bf b1 = load_bT_frag(xtb, C_, kc, n1, lane);
            acc0 = wmma_bf16(a, b0, acc0);
            acc1 = wmma_bf16(a, b1, acc1);
        }
#pragma unroll
        for (int v = 0; v < 8; ++v) {                  // online softmax, both tiles
            float f0 = acc0[v], f1 = acc1[v];
            float mn = fmaxf(m[v], fmaxf(f0, f1));
            s[v] = s[v] * __expf(m[v] - mn) + __expf(f0 - mn) + __expf(f1 - mn);
            m[v] = mn;
        }
    }
#pragma unroll
    for (int v = 0; v < 8; ++v) {                      // reduce across 16-lane halves
        float mv = m[v], sv = s[v];
#pragma unroll
        for (int off = 1; off < 16; off <<= 1) {
            float mo = __shfl_xor(mv, off, 32);
            float so = __shfl_xor(sv, off, 32);
            float mn = fmaxf(mv, mo);
            sv = sv * __expf(mv - mn) + so * __expf(mo - mn);
            mv = mn;
        }
        m[v] = mv; s[v] = sv;
    }
    __shared__ float redm[8][16], reds[8][16];
    if ((lane & 15) == 0) {
        int half = lane >> 4;
#pragma unroll
        for (int v = 0; v < 8; ++v) {
            redm[wave][v + 8 * half] = m[v];
            reds[wave][v + 8 * half] = s[v];
        }
    }
    __syncthreads();
    if (threadIdx.x < 16) {
        int r = threadIdx.x;
        float mv = redm[0][r], sv = reds[0][r];
#pragma unroll
        for (int wv = 1; wv < 8; ++wv) {
            float mo = redm[wv][r], so = reds[wv][r];
            float mn = fmaxf(mv, mo);
            sv = sv * __expf(mv - mn) + so * __expf(mo - mn);
            mv = mn;
        }
        cst[(size_t)b * HW + q0 + r] = mv + __logf(sv);
    }
}

// ---------------- fused attention: yt[p][cb] = sum_q exp(f[p,q]-c[q]) * g[cb][q] -----
// Double-buffered async LDS staging of the shared xt q-rows (8x reuse per block).
__global__ void k_attn(const ushortT* __restrict__ xt, const ushortT* __restrict__ g,
                       const float* __restrict__ cst, float* __restrict__ yt) {
    int b = blockIdx.y;
    int p0 = blockIdx.x * 128;
    int tid = threadIdx.x;
    int wave = tid >> 5, lane = tid & 31;
    const ushortT* xtb = xt + (size_t)b * HW * C_;
    const ushortT* gb  = g  + (size_t)b * CB * HW;
    const float* cstb  = cst + (size_t)b * HW;
    float* ytb = yt + (size_t)b * HW * CB;

    __shared__ __align__(16) ushortT Xq[2][32][XLD];   // double-buffered B tile
    __shared__ __align__(16) ushortT Psh[128][PLD];    // P' = exp(F - c[q]) tile

    v8f yacc[8];
#pragma unroll
    for (int t = 0; t < 8; ++t) yacc[t] = (v8f){};

    int myrow0 = p0 + wave * 16;
    int col = lane & 15, half = lane >> 4;
    int n0 = wave * 16;                                // this wave's cb tile (y phase)

    fill_xq(xtb, 0, &Xq[0][0][0], tid);
    wait_async0();
    __syncthreads();

    for (int qi = 0; qi < HW / 32; ++qi) {
        int q0 = qi * 32;
        int cur = qi & 1;
        if (q0 + 32 < HW)                              // prefetch next chunk (async)
            fill_xq(xtb, q0 + 32, &Xq[cur ^ 1][0][0], tid);
        __builtin_prefetch(gb + (size_t)(n0 + col) * HW + (q0 + 32 < HW ? q0 + 32 : q0), 0, 1);

        // ---- F phase: F[p, q] 16x32 tile per wave, K = C = 256, B from LDS ----
        v8f f0 = {}, f1 = {};
#pragma unroll
        for (int kc = 0; kc < C_; kc += 32) {
            v16bf a  = load_a_frag(xtb, C_, myrow0, kc, lane);
            v16bf b0 = load_bT_frag(&Xq[cur][0][0], XLD, kc, 0, lane);
            v16bf b1 = load_bT_frag(&Xq[cur][0][0], XLD, kc, 16, lane);
            f0 = wmma_bf16(a, b0, f0);
            f1 = wmma_bf16(a, b1, f1);
        }
        float c0v = cstb[q0 + col];
        float c1v = cstb[q0 + 16 + col];
        int rtile = wave * 16 + 8 * half;
#pragma unroll
        for (int v = 0; v < 8; ++v) {
            Psh[rtile + v][col]      = f2bf(__expf(f0[v] - c0v));
            Psh[rtile + v][16 + col] = f2bf(__expf(f1[v] - c1v));
        }
        __syncthreads();
        // ---- y phase: yacc[t] += P'[t*16.., 0..31] @ g[cb, q0..q0+31]^T ----
        v16bf bG = load_bT_frag(gb, HW, q0, n0, lane);
#pragma unroll
        for (int t = 0; t < 8; ++t) {
            v16bf aP = load_a_frag(&Psh[0][0], PLD, t * 16, 0, lane);
            yacc[t] = wmma_bf16(aP, bG, yacc[t]);
        }
        wait_async0();                                 // next Xq buffer landed
        __syncthreads();
    }
#pragma unroll
    for (int t = 0; t < 8; ++t)
#pragma unroll
        for (int v = 0; v < 8; ++v) {
            int p = p0 + t * 16 + v + 8 * half;
            ytb[(size_t)p * CB + n0 + col] = yacc[t][v];
        }
}

// ---------------- final: out = BN(Wz @ y + bz) --------------------------------------
__global__ void k_final(const float* __restrict__ yt, const float* __restrict__ Wz,
                        const float* __restrict__ bz, const float* __restrict__ gamma,
                        const float* __restrict__ beta, const float* __restrict__ rmean,
                        const float* __restrict__ rvar, float* __restrict__ out) {
    int b  = blockIdx.z;
    int co0 = blockIdx.y * 4;
    int p  = blockIdx.x * 256 + threadIdx.x;
    __shared__ float wrow[4][CB];
    for (int i = threadIdx.x; i < 4 * CB; i += blockDim.x)
        wrow[i / CB][i % CB] = Wz[(size_t)(co0 + i / CB) * CB + (i % CB)];
    __syncthreads();
    const float* yrow = yt + ((size_t)b * HW + p) * CB;
    float acc[4] = {0.f, 0.f, 0.f, 0.f};
#pragma unroll 4
    for (int cb = 0; cb < CB; ++cb) {
        float yv = yrow[cb];
        acc[0] += wrow[0][cb] * yv;
        acc[1] += wrow[1][cb] * yv;
        acc[2] += wrow[2][cb] * yv;
        acc[3] += wrow[3][cb] * yv;
    }
#pragma unroll
    for (int j = 0; j < 4; ++j) {
        int co = co0 + j;
        float inv = gamma[co] * rsqrtf(rvar[co] + 1e-5f);
        out[((size_t)b * C_ + co) * HW + p] = (acc[j] + bz[co] - rmean[co]) * inv + beta[co];
    }
}

extern "C" void kernel_launch(void* const* d_in, const int* in_sizes, int n_in,
                              void* d_out, int out_size, void* d_ws, size_t ws_size,
                              hipStream_t stream) {
    const float* x     = (const float*)d_in[0];
    const float* Wg    = (const float*)d_in[1];
    const float* bg    = (const float*)d_in[2];
    const float* Wz    = (const float*)d_in[3];
    const float* bz    = (const float*)d_in[4];
    const float* gamma = (const float*)d_in[5];
    const float* beta  = (const float*)d_in[6];
    const float* rmean = (const float*)d_in[7];
    const float* rvar  = (const float*)d_in[8];

    char* ws = (char*)d_ws;
    ushortT* xt  = (ushortT*)ws;                                   // 8 MiB
    ushortT* g   = (ushortT*)(ws + (8u  << 20));                   // 4 MiB
    ushortT* wgb = (ushortT*)(ws + (12u << 20));                   // 64 KiB
    float*   cst = (float*)  (ws + (12u << 20) + (64u  << 10));    // 64 KiB
    float*   yt  = (float*)  (ws + (12u << 20) + (128u << 10));    // 8 MiB
    float*   out = (float*)d_out;

    k_xt   <<<dim3(HW / 32, C_ / 32, NB), dim3(32, 8), 0, stream>>>(x, xt);
    k_cvt  <<<(CB * C_) / 256, 256, 0, stream>>>(Wg, wgb, CB * C_);
    k_g    <<<dim3(HW / 32, NB), 256, 0, stream>>>(xt, wgb, bg, g);
    k_stats<<<dim3(HW / 16, NB), 256, 0, stream>>>(xt, cst);
    k_attn <<<dim3(HW / 128, NB), 256, 0, stream>>>(xt, g, cst, yt);
    k_final<<<dim3(HW / 256, C_ / 4, NB), 256, 0, stream>>>(yt, Wz, bz, gamma, beta, rmean, rvar, out);
}
